// LIFNeuronLayer_47674136985991
// MI455X (gfx1250) — compile-verified, matrix operand order checked by reading, not run on hardware
//
#include <hip/hip_runtime.h>
#include <cstdint>
#include <cstddef>

#define BATCH  128
#define IDIM   2048
#define ODIM   2048
#define TSTEPS 100
#define NKCH   (IDIM / 32)     // 64 K-chunks of 32
#define NREG   32              // K-chunks with B resident in VGPRs (K 0..1023)
#define AROW   (IDIM + 8)      // padded LDS row for A (halfs): 4-bank skew
#define BUFH   (16 * AROW)     // one A slice buffer (halfs)
#define WBROW  (1024 + 8)      // padded LDS row for B-half (halfs)

typedef __attribute__((ext_vector_type(16))) __bf16 v16bf;
typedef __attribute__((ext_vector_type(8)))  float  v8f;

// round-to-nearest-even float -> bf16 (raw bits)
__device__ __forceinline__ uint16_t f2bf(float f) {
    unsigned u = __builtin_bit_cast(unsigned, f);
    u += 0x7FFFu + ((u >> 16) & 1u);
    return (uint16_t)(u >> 16);
}

// load a 16-element bf16 fragment (32 bytes) as two b128 loads
__device__ __forceinline__ v16bf load_frag16(const uint16_t* p) {
    union { uint4 q[2]; v16bf v; } u;
    u.q[0] = *(const uint4*)(p);
    u.q[1] = *(const uint4*)(p + 8);
    return u.v;
}

// async 16B copy global -> LDS (gfx1250, ASYNCcnt-tracked, GVS addressing)
__device__ __forceinline__ void async_copy16(uint32_t lds_off, uint32_t goff,
                                             const uint16_t* gbase) {
    asm volatile("global_load_async_to_lds_b128 %0, %1, %2"
                 :
                 : "v"(lds_off), "v"(goff), "s"(gbase)
                 : "memory");
}
__device__ __forceinline__ void wait_async0() {
    asm volatile("s_wait_asynccnt 0x0" ::: "memory");
}

// ---------------------------------------------------------------------------
// Kernel 1: pack spikes [B,I,T] f32  ->  [T,B,I] bf16 (LDS transpose)
// ---------------------------------------------------------------------------
__global__ __launch_bounds__(256) void pack_spikes(const float* __restrict__ sp,
                                                   uint16_t* __restrict__ spT) {
    __shared__ float ldsf[128 * TSTEPS];           // 51200 B
    const int i0 = blockIdx.x * 128;
    const int b  = blockIdx.y;
    const size_t inbase = ((size_t)b * IDIM + i0) * TSTEPS;
    for (int idx = threadIdx.x; idx < 128 * TSTEPS; idx += 256)
        ldsf[idx] = sp[inbase + idx];
    __syncthreads();
    for (int odx = threadIdx.x; odx < 128 * TSTEPS; odx += 256) {
        int t  = odx >> 7;       // 0..99
        int ii = odx & 127;
        spT[(size_t)t * (BATCH * IDIM) + (size_t)b * IDIM + i0 + ii] =
            f2bf(ldsf[ii * TSTEPS + t]);
    }
}

// ---------------------------------------------------------------------------
// Kernel 2: weight [I,O] f32 -> Wt [O,I] bf16 (tiled LDS transpose)
// ---------------------------------------------------------------------------
__global__ __launch_bounds__(256) void transpose_weight(const float* __restrict__ w,
                                                        uint16_t* __restrict__ wT) {
    __shared__ float tile[32][33];
    const int o0 = blockIdx.x * 32;
    const int i0 = blockIdx.y * 32;
    const int tx = threadIdx.x;    // 0..31
    const int ty = threadIdx.y;    // 0..7
    for (int r = ty; r < 32; r += 8)
        tile[r][tx] = w[(size_t)(i0 + r) * ODIM + o0 + tx];
    __syncthreads();
    for (int r = ty; r < 32; r += 8)
        wT[(size_t)(o0 + r) * IDIM + i0 + tx] = f2bf(tile[tx][r]);
}

// ---------------------------------------------------------------------------
// Kernel 3: LIF layer (hybrid-stationary).
//   - B chunks 0..31  (K 0..1023):   resident in 256 VGPRs for whole kernel
//   - B chunks 32..63 (K 1024..2047): resident in LDS (132 KB), staged once
//   - A slice (16 x 2048 bf16): LDS double-buffer, async-prefetched one full
//     timestep ahead (global_load_async_to_lds_b128 / s_wait_asynccnt)
//   - plain load->wmma body (no manual rotation) so the scheduler batches
//     ds_loads ahead of the 64-deep v_wmma_f32_16x16x32_bf16 chain
// ---------------------------------------------------------------------------
__global__ __launch_bounds__(128) void lif_main(const uint16_t* __restrict__ spT,
                                                const uint16_t* __restrict__ wT,
                                                float* __restrict__ out) {
    extern __shared__ uint16_t smem[];   // [2*BUFH A halfs][64*WBROW B halfs]
    uint16_t* lds_wb = smem + 2 * BUFH;

    const int tid   = threadIdx.x;
    const int lane  = tid & 31;
    const int wv    = tid >> 5;                    // wave 0..3 -> 16 O-cols each
    const int ocol0 = blockIdx.x * 64;
    const int btile = blockIdx.y * 16;
    const int koff  = (lane >> 4) * 16;            // contiguous K-split: lane half h holds k 16h..16h+15

    const uint32_t lds_base = (uint32_t)(uintptr_t)smem;   // LDS byte offset

    // ---- stage LDS-resident B half (K 1024..2047), async, once ----
    {
        const uint32_t lwb = lds_base + 2 * BUFH * 2;
        #pragma unroll
        for (int c = 0; c < 64; ++c) {             // 8192 16B chunks / 128 thr
            int idx = tid + c * 128;
            int r   = idx >> 7;                    // 0..63  (local o column)
            int col = (idx & 127) * 8;             // halfs within K-half
            async_copy16(lwb + (uint32_t)(r * WBROW + col) * 2,
                         (uint32_t)(r * IDIM + 1024 + col) * 2,
                         wT + (size_t)ocol0 * IDIM);
        }
    }

    // ---- register-resident B half (K 0..1023) ----
    const uint16_t* wbase =
        wT + (size_t)(ocol0 + wv * 16 + (lane & 15)) * IDIM + koff;
    v16bf breg[NREG];
    #pragma unroll
    for (int kc = 0; kc < NREG; ++kc)
        breg[kc] = load_frag16(wbase + kc * 32);

    // cooperative async stage of one 16 x 2048-half A slice into buffer `buf`
    auto stage = [&](int buf, int t) {
        const uint16_t* gbase =
            spT + (size_t)t * (BATCH * IDIM) + (size_t)btile * IDIM;
        const uint32_t lbuf = lds_base + (uint32_t)buf * (BUFH * 2);
        #pragma unroll
        for (int c = 0; c < 32; ++c) {             // 4096 16B chunks / 128 thr
            int idx = tid + c * 128;
            int r   = idx >> 8;                    // 0..15
            int col = (idx & 255) * 8;             // halfs
            async_copy16(lbuf + (uint32_t)(r * AROW + col) * 2,
                         (uint32_t)(r * IDIM + col) * 2, gbase);
        }
    };

    // prologue: A slice for t=0 (weight-half async still in flight too)
    stage(0, 0);
    wait_async0();
    __syncthreads();

    const int o     = ocol0 + wv * 16 + (lane & 15);
    const int mbase = (lane < 16) ? 0 : 8;         // C/D layout: M = j (+8)
    const uint16_t* bbase = lds_wb + (wv * 16 + (lane & 15)) * WBROW + koff;

    v8f m = {0.f, 0.f, 0.f, 0.f, 0.f, 0.f, 0.f, 0.f};   // membrane

    for (int t = 0; t < TSTEPS; ++t) {
        const int p = t & 1;
        if (t + 1 < TSTEPS) stage(1 - p, t + 1);   // hidden under 64 WMMAs

        const uint16_t* fbase = smem + p * BUFH + (lane & 15) * AROW + koff;

        v8f acc = m;                               // membrane folds into C
        #pragma unroll
        for (int kc = 0; kc < NREG; ++kc) {        // B from registers
            v16bf a = load_frag16(fbase + kc * 32);
            acc = __builtin_amdgcn_wmma_f32_16x16x32_bf16(
                    false, a, false, breg[kc], (short)0, acc, false, false);
        }
        #pragma unroll
        for (int kc = NREG; kc < NKCH; ++kc) {     // B from LDS
            v16bf a = load_frag16(fbase + kc * 32);
            v16bf b = load_frag16(bbase + (kc - NREG) * 32);
            acc = __builtin_amdgcn_wmma_f32_16x16x32_bf16(
                    false, a, false, b, (short)0, acc, false, false);
        }

        // ---- LIF threshold / reset / store fires ----
        #pragma unroll
        for (int j = 0; j < 8; ++j) {
            float v     = acc[j];
            float fired = (v > 1.0f) ? 1.0f : 0.0f;
            int   bb    = btile + mbase + j;
            out[((size_t)bb * ODIM + o) * TSTEPS + t] = fired;
            m[j] = (v > 1.0f) ? 0.0f : v;          // reset-to-zero
        }

        wait_async0();                             // next A slice landed
        __syncthreads();                           // all waves done with buf p
    }
}

// ---------------------------------------------------------------------------
extern "C" void kernel_launch(void* const* d_in, const int* in_sizes, int n_in,
                              void* d_out, int out_size, void* d_ws, size_t ws_size,
                              hipStream_t stream) {
    (void)in_sizes; (void)n_in; (void)out_size; (void)ws_size;
    const float* sp = (const float*)d_in[0];     // [B,I,T] f32
    const float* w  = (const float*)d_in[1];     // [I,O]   f32
    float* out      = (float*)d_out;             // [B,O,T] f32

    uint16_t* spT = (uint16_t*)d_ws;                                      // 52.4 MB
    uint16_t* wT  = (uint16_t*)((char*)d_ws +
                     (size_t)TSTEPS * BATCH * IDIM * sizeof(uint16_t));   // +8.4 MB

    pack_spikes<<<dim3(IDIM / 128, BATCH), 256, 0, stream>>>(sp, spT);
    transpose_weight<<<dim3(ODIM / 32, IDIM / 32), dim3(32, 8), 0, stream>>>(w, wT);

    const size_t shmem = (size_t)(2 * BUFH + 64 * WBROW) * sizeof(uint16_t); // 263680 B
    lif_main<<<dim3(ODIM / 64, BATCH / 16), 128, shmem, stream>>>(spT, wT, out);
}
